// VQCodebook_53249004535975
// MI455X (gfx1250) — compile-verified, matrix operand order checked by reading, not run on hardware
//
#include <hip/hip_runtime.h>
#include <hip/hip_bf16.h>

// ---------------------------------------------------------------------------
// VQ codebook step for MI455X (gfx1250): wave32, WMMA bf16 16x16x32,
// Tensor Data Mover double-buffered LDS staging.
// ---------------------------------------------------------------------------

#define DIM     256
#define KCODES  8192
#define NVEC    32768
#define BETA    0.25f
#define DECAY   0.99f
#define EPSV    1e-5f

#define KC             64      // codes per chunk (4 groups of 16)
#define NCHUNK         (KCODES / KC)                     // 128
#define ROWS_PER_WAVE  16
#define WAVES_PER_BLK  8
#define ROWS_PER_BLK   (ROWS_PER_WAVE * WAVES_PER_BLK)   // 128

// packed E chunk: [Eh 64x256 bf16 | El 64x256 bf16] = 65536 bytes
#define CHUNK_BYTES    (KC * DIM * 2 * 2)
#define CHUNK_ELEMS    (KC * DIM * 2)

// LDS layout (bytes): buf0 @0, buf1 @65536, zh tile @131072, zl tile @196608
#define LDS_BUF0       0
#define LDS_BUF1       CHUNK_BYTES
#define LDS_ZH         (2 * CHUNK_BYTES)
#define LDS_ZL         (2 * CHUNK_BYTES + ROWS_PER_BLK * DIM * 2)
#define LDS_TOTAL      (2 * CHUNK_BYTES + 2 * ROWS_PER_BLK * DIM * 2)  // 256 KB

typedef __attribute__((ext_vector_type(16))) __bf16 v16bf;
typedef __attribute__((ext_vector_type(8)))  float  v8f;
typedef unsigned int tdm_v4u __attribute__((ext_vector_type(4)));
typedef int          tdm_v8i __attribute__((ext_vector_type(8)));
typedef int          tdm_v4i __attribute__((ext_vector_type(4)));

#if __has_builtin(__builtin_amdgcn_tensor_load_to_lds)
#define HAVE_TDM 1
#if __has_include(<hip/amd_detail/amd_gfx1250_TDM.h>)
#define TDM_SIX_ARGS 1
#endif
#endif

union FragU { v16bf v; uint4 q[2]; };

// 16-element bf16 fragment per documented 16-bit A/B VGPR layout:
// lanes 0-15 take depth {0..7,16..23}, lanes 16-31 take {8..15,24..31}.
__device__ __forceinline__ v16bf load_frag(const __bf16* rowPtr, int half) {
    FragU f;
    const char* p = (const char*)rowPtr;
    f.q[0] = *(const uint4*)(p + (size_t)(half * 8) * 2);
    f.q[1] = *(const uint4*)(p + (size_t)(16 + half * 8) * 2);
    return f.v;
}

#if defined(HAVE_TDM)
// Issue one TDM load: 64KB packed E chunk (1-D tile of 8192 x 8-byte elems)
// from global `gaddr` into LDS byte offset `lds_off`.  Tracked by TENSORcnt.
__device__ __forceinline__ void tdm_load_chunk(const void* gaddr, unsigned lds_off) {
    const unsigned long long ga = (unsigned long long)gaddr;
    tdm_v4u g0;
    g0.x = 1u;                                       // count=1, user descriptor
    g0.y = lds_off;                                  // lds_addr (bytes)
    g0.z = (unsigned)(ga & 0xffffffffu);             // global_addr[31:0]
    g0.w = (unsigned)((ga >> 32) & 0x01ffffffu)      // global_addr[56:32]
         | 0x80000000u;                              // type=2 ("image")
    // group1: data_size=8B; tensor_dim0=0x2000; tensor_dim1=1;
    //         tile_dim0=0x2000; tile_dim1=1; dim0_stride=0x2000
    tdm_v8i g1 = { 0x00030000, 0x20000000, 0x00010000, 0x20000000,
                   0x00000001, 0x00002000, 0, 0 };
    tdm_v4i g2 = { 0, 0, 0, 0 };
    tdm_v4i g3 = { 0, 0, 0, 0 };
#if defined(TDM_SIX_ARGS)
    tdm_v8i gx = { 0, 0, 0, 0, 0, 0, 0, 0 };
    __builtin_amdgcn_tensor_load_to_lds(g0, g1, g2, g3, gx, 0);
#else
    __builtin_amdgcn_tensor_load_to_lds(g0, g1, g2, g3, 0);
#endif
}
#endif

// ---------------------------------------------------------------------------
// K0: split embeddings into bf16 hi/lo (chunk-packed layout), e2[k]=sum(E^2)
// ---------------------------------------------------------------------------
__global__ void k0_prep_E(const float* __restrict__ E,
                          __bf16* __restrict__ Epk, float* __restrict__ e2) {
    __shared__ float red[256];
    const int k = blockIdx.x, d = threadIdx.x;
    const float v = E[(size_t)k * DIM + d];
    const __bf16 h = (__bf16)v;
    const __bf16 l = (__bf16)(v - (float)h);
    const int chunk = k >> 6, w = k & 63;
    const size_t base = (size_t)chunk * CHUNK_ELEMS;
    Epk[base + (size_t)w * DIM + d]            = h;   // hi half of chunk
    Epk[base + KC * DIM + (size_t)w * DIM + d] = l;   // lo half of chunk
    red[d] = v * v;
    __syncthreads();
    for (int s = 128; s > 0; s >>= 1) {
        if (d < s) red[d] += red[d + s];
        __syncthreads();
    }
    if (d == 0) e2[k] = red[0];
}

// ---------------------------------------------------------------------------
// K1: split z into bf16 hi/lo
// ---------------------------------------------------------------------------
__global__ void k1_prep_z(const float* __restrict__ z,
                          __bf16* __restrict__ zh, __bf16* __restrict__ zl) {
    const size_t i = (size_t)blockIdx.x * DIM + threadIdx.x;
    const float v = z[i];
    const __bf16 h = (__bf16)v;
    zh[i] = h;
    zl[i] = (__bf16)(v - (float)h);
}

// ---------------------------------------------------------------------------
// compute one 64-code chunk: 8 depth steps x 4 groups x 3 compensated WMMAs,
// then fold e2 - 2*acc into the per-lane running argmin.
// ---------------------------------------------------------------------------
__device__ __forceinline__ void compute_chunk(const __bf16* __restrict__ ehs,
                                              const __bf16* __restrict__ els,
                                              const __bf16* __restrict__ zrh,
                                              const __bf16* __restrict__ zrl,
                                              const float*  __restrict__ e2,
                                              int kk0, int half, int col,
                                              float* best, int* bidx) {
    v8f acc[4];
#pragma unroll
    for (int g = 0; g < 4; ++g) acc[g] = (v8f){};

#pragma unroll
    for (int c = 0; c < 8; ++c) {
        const v16bf ah = load_frag(zrh + c * 32, half);
        const v16bf al = load_frag(zrl + c * 32, half);
#pragma unroll
        for (int g = 0; g < 4; ++g) {
            const v16bf bh = load_frag(ehs + (size_t)(g * 16 + col) * DIM + c * 32, half);
            const v16bf bl = load_frag(els + (size_t)(g * 16 + col) * DIM + c * 32, half);
            // compensated product: zh*eh + zh*el + zl*eh (~fp32 accurate)
            acc[g] = __builtin_amdgcn_wmma_f32_16x16x32_bf16(false, ah, false, bh,
                                                             (short)0, acc[g], false, false);
            acc[g] = __builtin_amdgcn_wmma_f32_16x16x32_bf16(false, ah, false, bl,
                                                             (short)0, acc[g], false, false);
            acc[g] = __builtin_amdgcn_wmma_f32_16x16x32_bf16(false, al, false, bh,
                                                             (short)0, acc[g], false, false);
        }
    }
    // C layout: VGPR j, half h -> row (j + 8*h), col (lane&15)
#pragma unroll
    for (int g = 0; g < 4; ++g) {
        const int kbase = kk0 + g * 16;
        const float e2v  = e2[kbase + col];
        const int   myId = kbase + col;
#pragma unroll
        for (int j = 0; j < 8; ++j) {
            const float d2 = e2v - 2.0f * acc[g][j];
            if (d2 < best[j]) { best[j] = d2; bidx[j] = myId; }
        }
    }
}

// ---------------------------------------------------------------------------
// K2: fused GEMM (z @ E^T, compensated bf16 WMMA) + per-row argmin.
//   - z tile (128 rows, hi+lo) staged in LDS once
//   - E chunks streamed by the Tensor Data Mover into a double-buffered LDS
//     slot, overlapped with compute (TENSORcnt / s_wait_tensorcnt)
// ---------------------------------------------------------------------------
__global__ void k2_gemm_argmin(const __bf16* __restrict__ zh,
                               const __bf16* __restrict__ zl,
                               const __bf16* __restrict__ Epk,
                               const float*  __restrict__ e2,
                               int*   __restrict__ idx_ws,
                               float* __restrict__ out_idx) {
    extern __shared__ char smem[];
    __bf16* zhs = (__bf16*)(smem + LDS_ZH);
    __bf16* zls = (__bf16*)(smem + LDS_ZL);

    const int tid  = threadIdx.x;
    const int lane = tid & 31;
    const int wave = tid >> 5;
    const int half = lane >> 4;
    const int col  = lane & 15;
    const int rowBase = blockIdx.x * ROWS_PER_BLK + wave * ROWS_PER_WAVE;
    const char* epkb = (const char*)Epk;

#if defined(HAVE_TDM)
    if (wave == 0) {   // prologue: prime both LDS buffers via the TDM
        tdm_load_chunk(epkb, LDS_BUF0);
        tdm_load_chunk(epkb + CHUNK_BYTES, LDS_BUF1);
    }
#endif

    // ---- stage the block's z tile (hi+lo) into LDS once ----
    {
        const uint4* srcH = (const uint4*)(zh + (size_t)blockIdx.x * ROWS_PER_BLK * DIM);
        const uint4* srcL = (const uint4*)(zl + (size_t)blockIdx.x * ROWS_PER_BLK * DIM);
        uint4* dstH = (uint4*)zhs;
        uint4* dstL = (uint4*)zls;
        const int total = ROWS_PER_BLK * DIM / 8;
        for (int i = tid; i < total; i += blockDim.x) {
            dstH[i] = srcH[i];
            dstL[i] = srcL[i];
        }
    }

    const __bf16* zrh = zhs + (size_t)(wave * ROWS_PER_WAVE + col) * DIM;
    const __bf16* zrl = zls + (size_t)(wave * ROWS_PER_WAVE + col) * DIM;

    float best[8];
    int   bidx[8];
#pragma unroll
    for (int j = 0; j < 8; ++j) { best[j] = 3.4e38f; bidx[j] = 0; }

#if defined(HAVE_TDM)
    for (int ci = 0; ci < NCHUNK; ++ci) {
        // wave 0 has the outstanding TDM ops; others pass through instantly
        if (ci + 1 < NCHUNK) __builtin_amdgcn_s_wait_tensorcnt(1);
        else                 __builtin_amdgcn_s_wait_tensorcnt(0);
        __syncthreads();   // chunk ci resident in buf[ci&1] for all waves

        const __bf16* ehs = (const __bf16*)(smem + (ci & 1) * CHUNK_BYTES);
        const __bf16* els = ehs + KC * DIM;
        compute_chunk(ehs, els, zrh, zrl, e2, ci * KC, half, col, best, bidx);

        __syncthreads();   // all waves done with buf[ci&1]
        if (wave == 0 && ci + 2 < NCHUNK)
            tdm_load_chunk(epkb + (size_t)(ci + 2) * CHUNK_BYTES,
                           (unsigned)((ci & 1) * CHUNK_BYTES));
    }
#else
    // fallback: synchronous cooperative staging (single buffer)
    for (int ci = 0; ci < NCHUNK; ++ci) {
        __syncthreads();
        {
            const uint4* src = (const uint4*)(epkb + (size_t)ci * CHUNK_BYTES);
            uint4* dst = (uint4*)smem;
            const int total = CHUNK_BYTES / 16;
            for (int i = tid; i < total; i += blockDim.x) dst[i] = src[i];
        }
        __syncthreads();
        const __bf16* ehs = (const __bf16*)smem;
        const __bf16* els = ehs + KC * DIM;
        compute_chunk(ehs, els, zrh, zrl, e2, ci * KC, half, col, best, bidx);
    }
#endif

    // reduce argmin across the 16 lanes of each half, lowest index on ties
#pragma unroll
    for (int j = 0; j < 8; ++j) {
        float bv = best[j];
        int   bi = bidx[j];
#pragma unroll
        for (int m = 1; m <= 8; m <<= 1) {
            const float ov = __shfl_xor(bv, m, 32);
            const int   oi = __shfl_xor(bi, m, 32);
            if (ov < bv || (ov == bv && oi < bi)) { bv = ov; bi = oi; }
        }
        if (col == 0) {
            const int r = rowBase + half * 8 + j;
            idx_ws[r]  = bi;
            out_idx[r] = (float)bi;
        }
    }
}

// ---------------------------------------------------------------------------
// K3: gather q = E[idx] (quantized_st == q numerically), scatter segment
//     sums with atomics, block-reduce the loss.
// ---------------------------------------------------------------------------
__global__ void k3_gather_scatter(const float* __restrict__ z,
                                  const float* __restrict__ E,
                                  const int*   __restrict__ idx,
                                  float* __restrict__ out_q,
                                  float* __restrict__ counts,
                                  float* __restrict__ embed_sum,
                                  float* __restrict__ loss_acc) {
    __shared__ float red[256];
    const int i = blockIdx.x, d = threadIdx.x;
    const int k = idx[i];
    const float zv = z[(size_t)i * DIM + d];
    const float qv = E[(size_t)k * DIM + d];
    out_q[(size_t)i * DIM + d] = qv;          // z + (q - z) == q
    atomicAdd(&embed_sum[(size_t)k * DIM + d], zv);
    if (d == 0) atomicAdd(&counts[k], 1.0f);
    const float diff = zv - qv;
    red[d] = diff * diff;
    __syncthreads();
    for (int s = 128; s > 0; s >>= 1) {
        if (d < s) red[d] += red[d + s];
        __syncthreads();
    }
    if (d == 0) atomicAdd(loss_acc, red[0]);
}

// ---------------------------------------------------------------------------
// K4a: new_cluster_size + usage, reduce n = sum(new_cluster_size)
// ---------------------------------------------------------------------------
__global__ void k4a_cluster(const float* __restrict__ ecs,
                            const float* __restrict__ counts,
                            const int*   __restrict__ usage,
                            float* __restrict__ out_ncs,
                            float* __restrict__ out_usage,
                            float* __restrict__ ncs_ws,
                            float* __restrict__ n_acc) {
    __shared__ float red[256];
    const int t = threadIdx.x;
    const int k = blockIdx.x * 256 + t;
    const float c   = counts[k];
    const float ncs = DECAY * ecs[k] + (1.0f - DECAY) * c;
    out_ncs[k]  = ncs;
    ncs_ws[k]   = ncs;
    out_usage[k] = (float)usage[k] + c;
    red[t] = ncs;
    __syncthreads();
    for (int s = 128; s > 0; s >>= 1) {
        if (t < s) red[t] += red[t + s];
        __syncthreads();
    }
    if (t == 0) atomicAdd(n_acc, red[0]);
}

// ---------------------------------------------------------------------------
// K4b: new_embed_sum, normalized new_embeddings, final loss scalar
// ---------------------------------------------------------------------------
__global__ void k4b_embed(const float* __restrict__ ees,
                          const float* __restrict__ embed_sum,
                          const float* __restrict__ ncs_ws,
                          const float* __restrict__ n_acc,
                          const float* __restrict__ loss_acc,
                          float* __restrict__ out_nes,
                          float* __restrict__ out_emb,
                          float* __restrict__ out_loss) {
    const size_t flat = (size_t)blockIdx.x * 256 + threadIdx.x;
    const int k = (int)(flat / DIM);
    const float nes = DECAY * ees[flat] + (1.0f - DECAY) * embed_sum[flat];
    out_nes[flat] = nes;
    const float n = *n_acc;
    const float cluster = (ncs_ws[k] + EPSV) / (n + (float)KCODES * EPSV) * n;
    out_emb[flat] = nes / cluster;
    if (flat == 0)
        out_loss[0] = (1.0f + BETA) * loss_acc[0] / ((float)NVEC * (float)DIM);
}

// ---------------------------------------------------------------------------
// launcher
// ---------------------------------------------------------------------------
extern "C" void kernel_launch(void* const* d_in, const int* in_sizes, int n_in,
                              void* d_out, int out_size, void* d_ws, size_t ws_size,
                              hipStream_t stream) {
    (void)in_sizes; (void)n_in; (void)out_size; (void)ws_size;

    const float* z    = (const float*)d_in[0];
    const float* E    = (const float*)d_in[1];
    const float* ecs  = (const float*)d_in[2];
    const float* ees  = (const float*)d_in[3];
    const int*   usage= (const int*)  d_in[4];

    // workspace layout (bytes)
    constexpr size_t NB = (size_t)NVEC * DIM;
    constexpr size_t KB = (size_t)KCODES * DIM;
    constexpr size_t OFF_ZH   = 0;
    constexpr size_t OFF_ZL   = OFF_ZH  + NB * 2;
    constexpr size_t OFF_EPK  = OFF_ZL  + NB * 2;           // packed hi/lo chunks
    constexpr size_t OFF_E2   = OFF_EPK + KB * 2 * 2;
    constexpr size_t OFF_IDX  = OFF_E2  + (size_t)KCODES * 4;
    constexpr size_t OFF_CNT  = OFF_IDX + (size_t)NVEC * 4;
    constexpr size_t OFF_NCS  = OFF_CNT + (size_t)KCODES * 4;
    constexpr size_t OFF_SCAL = OFF_NCS + (size_t)KCODES * 4;  // [loss, n]
    constexpr size_t OFF_ESUM = OFF_SCAL + 256;

    char* ws = (char*)d_ws;
    __bf16* zh   = (__bf16*)(ws + OFF_ZH);
    __bf16* zl   = (__bf16*)(ws + OFF_ZL);
    __bf16* Epk  = (__bf16*)(ws + OFF_EPK);
    float*  e2   = (float*) (ws + OFF_E2);
    int*    idxw = (int*)   (ws + OFF_IDX);
    float*  cnts = (float*) (ws + OFF_CNT);
    float*  ncsw = (float*) (ws + OFF_NCS);
    float*  scal = (float*) (ws + OFF_SCAL);   // scal[0]=loss, scal[1]=n
    float*  esum = (float*) (ws + OFF_ESUM);

    // output layout (float elements), tuple flattened in return order
    float* out = (float*)d_out;
    constexpr size_t O_IDX  = 0;
    constexpr size_t O_Q    = O_IDX  + NVEC;
    constexpr size_t O_LOSS = O_Q    + NB;
    constexpr size_t O_EMB  = O_LOSS + 1;
    constexpr size_t O_NCS  = O_EMB  + KB;
    constexpr size_t O_NES  = O_NCS  + KCODES;
    constexpr size_t O_USE  = O_NES  + KB;

    // zero the accumulators every call (ws is not re-poisoned between replays)
    hipMemsetAsync(cnts, 0, (size_t)KCODES * 4, stream);
    hipMemsetAsync(esum, 0, KB * 4, stream);
    hipMemsetAsync(scal, 0, 2 * sizeof(float), stream);

    k0_prep_E<<<KCODES, 256, 0, stream>>>(E, Epk, e2);
    k1_prep_z<<<NVEC, 256, 0, stream>>>(z, zh, zl);

    k2_gemm_argmin<<<NVEC / ROWS_PER_BLK, 256, LDS_TOTAL, stream>>>(
        zh, zl, Epk, e2, idxw, out + O_IDX);

    k3_gather_scatter<<<NVEC, 256, 0, stream>>>(
        z, E, idxw, out + O_Q, cnts, esum, &scal[0]);

    k4a_cluster<<<KCODES / 256, 256, 0, stream>>>(
        ecs, cnts, usage, out + O_NCS, out + O_USE, ncsw, &scal[1]);

    k4b_embed<<<(unsigned)(KB / 256), 256, 0, stream>>>(
        ees, esum, ncsw, &scal[1], &scal[0],
        out + O_NES, out + O_EMB, out + O_LOSS);
}
